// GPTLanguageModel_66589172957723
// MI455X (gfx1250) — matrix operand order repaired and checked by reading, not compiled, not run on hardware
//
#include <hip/hip_runtime.h>
#include <hip/hip_bf16.h>
#include <math.h>

typedef __attribute__((ext_vector_type(16))) __bf16 v16bf;
typedef __attribute__((ext_vector_type(8)))  float  v8f;
typedef int v4i __attribute__((vector_size(16)));

namespace {
constexpr int kC  = 768;
constexpr int kH  = 12;
constexpr int kL  = 12;
constexpr int kB  = 2;
constexpr int kT  = 1024;
constexpr int kV  = 50257;
constexpr int kM  = kB * kT;     // 2048 tokens
constexpr int kD  = kC / kH;     // 64
constexpr int kC3 = 3 * kC;      // 2304
constexpr int kF  = 4 * kC;      // 3072
}

// ---------------------------------------------------------------------------
// Async global -> LDS copy (CDNA5 GLOBAL_LOAD_ASYNC_TO_LDS_B128, ASYNCcnt),
// with a synchronous fallback if the builtin is unavailable.
// Builtin signature (from compiler diagnostic): (v4i __device__*, ..., Ii, Ii)
// ---------------------------------------------------------------------------
#if defined(__gfx1250__) && __has_builtin(__builtin_amdgcn_global_load_async_to_lds_b128)
#define HAS_ASYNC_LDS 1
typedef __attribute__((address_space(1))) v4i* gas_ptr;
typedef __attribute__((address_space(3))) v4i* las_ptr;
#endif

__device__ inline void copy16_to_lds(const __bf16* g, __bf16* l) {
#ifdef HAS_ASYNC_LDS
  __builtin_amdgcn_global_load_async_to_lds_b128((gas_ptr)g, (las_ptr)l, 0, 0);
#else
  *reinterpret_cast<uint4*>(l) = *reinterpret_cast<const uint4*>(g);
#endif
}

__device__ inline void async_wait0() {
#ifdef HAS_ASYNC_LDS
#if __has_builtin(__builtin_amdgcn_s_wait_asynccnt)
  __builtin_amdgcn_s_wait_asynccnt(0);
#else
  asm volatile("s_wait_asynccnt 0" ::: "memory");
#endif
#endif
}

// ---------------------------------------------------------------------------
// WMMA fragment helpers (16-bit A 16x32 and B 32x16 layouts, wave32)
// A: lane L<16 holds row L, K {0..7,16..23}; lane L+16 holds row L, K {8..15,24..31}
// B (from LDS where p[col*ld + k]): lanes 0-15 -> K {0..7,16..23}, lanes 16-31 -> K {8..15,24..31}
// ---------------------------------------------------------------------------
__device__ inline v16bf frag_a_lds(const __bf16* p, int ld, int lane) {
  const int l16 = lane & 15;
  const __bf16* r = p + l16 * ld + ((lane & 16) ? 8 : 0);
  v16bf a;
#pragma unroll
  for (int i = 0; i < 4; ++i) {
    a[2 * i]     = r[2 * i];
    a[2 * i + 1] = r[2 * i + 1];
    a[8 + 2 * i] = r[16 + 2 * i];
    a[9 + 2 * i] = r[17 + 2 * i];
  }
  return a;
}

__device__ inline v16bf frag_b_lds(const __bf16* p, int ld, int lane) {
  const int l16 = lane & 15;
  const __bf16* r = p + l16 * ld + ((lane & 16) ? 8 : 0);
  v16bf b;
#pragma unroll
  for (int i = 0; i < 4; ++i) {
    b[2 * i]     = r[2 * i];
    b[2 * i + 1] = r[2 * i + 1];
    b[8 + 2 * i] = r[16 + 2 * i];
    b[9 + 2 * i] = r[17 + 2 * i];
  }
  return b;
}

__device__ inline v8f wmma_bf16(v16bf a, v16bf b, v8f c) {
  return __builtin_amdgcn_wmma_f32_16x16x32_bf16(false, a, false, b, (short)0, c,
                                                 false, false);
}

// ---------------------------------------------------------------------------
// bf16 WMMA GEMM with pre-transposed weights:
//   D[M,N] = act(A[M,K] @ BT[N,K]^T + bias) (+ resid)
// block = 256 threads (8 waves), tile 128x128, K-step 32,
// double-buffered LDS, async global->LDS staging.
// ---------------------------------------------------------------------------
template <int ACT, bool RESID>
__global__ __launch_bounds__(256) void k_gemm(const __bf16* __restrict__ A,
                                              const __bf16* __restrict__ BT,
                                              const float* __restrict__ bias,
                                              const float* __restrict__ resid,
                                              float* __restrict__ outf,
                                              __bf16* __restrict__ outb,
                                              int M, int N, int K) {
  constexpr int LD = 40;            // padded k-stride (bf16 elems)
  constexpr int TILE = 128 * LD;    // one 128x32 tile
  __shared__ __bf16 sA[2 * TILE];   // [row][k]
  __shared__ __bf16 sB[2 * TILE];   // [col][k]

  const int tid  = threadIdx.x;
  const int lane = tid & 31, wave = tid >> 5;
  const int bm = blockIdx.y * 128, bn = blockIdx.x * 128;
  const int wm = (wave & 3) * 32, wn = (wave >> 2) * 64;

  v8f acc[2][4] = {};

  // staging roles: thread copies 16 bf16 (2 x b128) of one row/col
  const int srow = tid >> 1, shalf = (tid & 1) * 16;

  const __bf16* gA0 = A + (size_t)(bm + srow) * K + shalf;
  const int bcol = bn + srow;
  const __bf16* gB0 = BT + (size_t)bcol * K + shalf;
  const bool bok = (bcol < N);

  auto stage = [&](int buf, int kk) {
    __bf16* dA = &sA[buf * TILE + srow * LD + shalf];
    copy16_to_lds(gA0 + kk, dA);
    copy16_to_lds(gA0 + kk + 8, dA + 8);
    __bf16* dB = &sB[buf * TILE + srow * LD + shalf];
    if (bok) {
      copy16_to_lds(gB0 + kk, dB);
      copy16_to_lds(gB0 + kk + 8, dB + 8);
    } else {
#pragma unroll
      for (int i = 0; i < 16; ++i) dB[i] = (__bf16)0.f;
    }
  };

  const int nk = K >> 5;
  stage(0, 0);

  for (int t = 0; t < nk; ++t) {
    async_wait0();          // our async copies into buf (t&1) are done
    __syncthreads();        // everyone's copies visible; prev compute done
    if (t + 1 < nk) stage((t + 1) & 1, (t + 1) << 5);

    const __bf16* cA = &sA[(t & 1) * TILE];
    const __bf16* cB = &sB[(t & 1) * TILE];
    v16bf af[2], bfr[4];
#pragma unroll
    for (int i = 0; i < 2; ++i) af[i] = frag_a_lds(cA + (wm + 16 * i) * LD, LD, lane);
#pragma unroll
    for (int j = 0; j < 4; ++j) bfr[j] = frag_b_lds(cB + (wn + 16 * j) * LD, LD, lane);
#pragma unroll
    for (int i = 0; i < 2; ++i)
#pragma unroll
      for (int j = 0; j < 4; ++j)
        acc[i][j] = wmma_bf16(af[i], bfr[j], acc[i][j]);
  }

  // ---- epilogue (C layout: VGPR e -> row e / e+8; lane%16 -> col) ----
  const int l16 = lane & 15;
  const int rof = (lane & 16) ? 8 : 0;
#pragma unroll
  for (int i = 0; i < 2; ++i) {
#pragma unroll
    for (int j = 0; j < 4; ++j) {
      const int col = bn + wn + 16 * j + l16;
      if (col >= N) continue;
      const float bv = bias ? bias[col] : 0.f;
#pragma unroll
      for (int e = 0; e < 8; ++e) {
        const int row = bm + wm + 16 * i + rof + e;
        if (row >= M) continue;
        float v = acc[i][j][e] + bv;
        if (ACT == 1) v = 0.5f * v * (1.f + erff(v * 0.70710678118654752f));
        if (RESID) v += resid[(size_t)row * N + col];
        if (outf) outf[(size_t)row * N + col] = v;
        if (outb) outb[(size_t)row * N + col] = (__bf16)v;
      }
    }
  }
}

// ---------------------------------------------------------------------------
// Flash-style causal attention with WMMA. grid = (B*H, T/128), block = 256.
// Wave w handles 16 queries; block cooperatively stages 32-key K/V chunks.
// ---------------------------------------------------------------------------
__global__ __launch_bounds__(256) void k_attn(const float* __restrict__ qkv,
                                              __bf16* __restrict__ y) {
  constexpr int KLD = 72, VLD = 40, PLD = 40;
  __shared__ __bf16 sK[32 * KLD];        // [key][d]
  __shared__ __bf16 sV[64 * VLD];        // [d][key]  (transposed)
  __shared__ __bf16 sP[8][16 * PLD];     // per-wave probs [row][key]

  const int bh = blockIdx.x;
  const int b = bh / kH, h = bh % kH;
  const int tid = threadIdx.x, lane = tid & 31, wave = tid >> 5;
  const int qbase = blockIdx.y * 128 + wave * 16;
  const int l16 = lane & 15;
  const int rof = (lane & 16) ? 8 : 0;
  const float sc = 0.125f; // 1/sqrt(64)

  // Q fragments (pre-scaled), two K-dim steps over d=0..63
  v16bf qa[2];
#pragma unroll
  for (int ds = 0; ds < 2; ++ds) {
    const float* r = qkv + (size_t)(b * kT + qbase + l16) * kC3 + h * kD + ds * 32 +
                     ((lane & 16) ? 8 : 0);
    v16bf a;
#pragma unroll
    for (int i = 0; i < 4; ++i) {
      a[2 * i]     = (__bf16)(r[2 * i] * sc);
      a[2 * i + 1] = (__bf16)(r[2 * i + 1] * sc);
      a[8 + 2 * i] = (__bf16)(r[16 + 2 * i] * sc);
      a[9 + 2 * i] = (__bf16)(r[17 + 2 * i] * sc);
    }
    qa[ds] = a;
  }

  v8f o[4] = {};
  float mrow[8], lrow[8];
#pragma unroll
  for (int e = 0; e < 8; ++e) { mrow[e] = -3.0e38f; lrow[e] = 0.f; }

  const int skey = tid >> 3, sdg = (tid & 7) * 8;
  const int nj = blockIdx.y * 4 + 4;

  for (int j = 0; j < nj; ++j) {
    __syncthreads();
    {
      const size_t tok = (size_t)(b * kT + j * 32 + skey);
      const float* kp = qkv + tok * kC3 + kC + h * kD + sdg;
      const float* vp = qkv + tok * kC3 + 2 * kC + h * kD + sdg;
#pragma unroll
      for (int d = 0; d < 8; ++d) {
        sK[skey * KLD + sdg + d] = (__bf16)kp[d];
        sV[(sdg + d) * VLD + skey] = (__bf16)vp[d];
      }
    }
    __syncthreads();
    if (j * 32 > qbase + 15) continue;   // causal: nothing for this wave

    // scores: S[16q x 32k] as two 16-col accumulators
    v8f s[2] = {};
#pragma unroll
    for (int sub = 0; sub < 2; ++sub)
#pragma unroll
      for (int ds = 0; ds < 2; ++ds) {
        v16bf kf = frag_b_lds(&sK[sub * 16 * KLD + ds * 32], KLD, lane);
        s[sub] = wmma_bf16(qa[ds], kf, s[sub]);
      }

    const bool diag = (j * 32 + 31 > qbase);
    float p[2][8];
#pragma unroll
    for (int e = 0; e < 8; ++e) {
      float s0 = s[0][e], s1 = s[1][e];
      if (diag) {
        const int row = qbase + rof + e;
        if (j * 32 + l16 > row)      s0 = -3.0e38f;
        if (j * 32 + 16 + l16 > row) s1 = -3.0e38f;
      }
      float cm = fmaxf(s0, s1);
#pragma unroll
      for (int off = 1; off < 16; off <<= 1) cm = fmaxf(cm, __shfl_xor(cm, off, 32));
      const float nm = fmaxf(mrow[e], cm);
      const float alpha = expf(mrow[e] - nm);
      const float p0 = expf(s0 - nm), p1 = expf(s1 - nm);
      float rs = p0 + p1;
#pragma unroll
      for (int off = 1; off < 16; off <<= 1) rs += __shfl_xor(rs, off, 32);
      lrow[e] = lrow[e] * alpha + rs;
      mrow[e] = nm;
#pragma unroll
      for (int dt = 0; dt < 4; ++dt) o[dt][e] *= alpha;
      p[0][e] = p0;
      p[1][e] = p1;
    }

    // P (C layout) -> LDS -> A-fragment layout
#pragma unroll
    for (int sub = 0; sub < 2; ++sub)
#pragma unroll
      for (int e = 0; e < 8; ++e)
        sP[wave][(rof + e) * PLD + sub * 16 + l16] = (__bf16)p[sub][e];
    asm volatile("s_wait_dscnt 0" ::: "memory");  // wave-local LDS RAW fence

    v16bf pf = frag_a_lds(&sP[wave][0], PLD, lane);
#pragma unroll
    for (int dt = 0; dt < 4; ++dt) {
      v16bf vf = frag_b_lds(&sV[dt * 16 * VLD], VLD, lane);
      o[dt] = wmma_bf16(pf, vf, o[dt]);
    }
  }

  // normalize and write y[tok, h*64 + d] as bf16
#pragma unroll
  for (int dt = 0; dt < 4; ++dt)
#pragma unroll
    for (int e = 0; e < 8; ++e) {
      const size_t idx =
          (size_t)(b * kT + qbase + rof + e) * kC + h * kD + dt * 16 + l16;
      y[idx] = (__bf16)(o[dt][e] / lrow[e]);
    }
}

// ---------------------------------------------------------------------------
// Reductions + pointwise kernels
// ---------------------------------------------------------------------------
__device__ inline float blk_sum(float v, float* buf) {
  __syncthreads();
#pragma unroll
  for (int off = 16; off; off >>= 1) v += __shfl_xor(v, off, 32);
  const int w = threadIdx.x >> 5, ln = threadIdx.x & 31;
  if (ln == 0) buf[w] = v;
  __syncthreads();
  if (w == 0) {
    float t = (ln < 8) ? buf[ln] : 0.f;
#pragma unroll
    for (int off = 4; off; off >>= 1) t += __shfl_xor(t, off, 32);
    if (ln == 0) buf[0] = t;
  }
  __syncthreads();
  return buf[0];
}

__device__ inline float blk_max(float v, float* buf) {
  __syncthreads();
#pragma unroll
  for (int off = 16; off; off >>= 1) v = fmaxf(v, __shfl_xor(v, off, 32));
  const int w = threadIdx.x >> 5, ln = threadIdx.x & 31;
  if (ln == 0) buf[w] = v;
  __syncthreads();
  if (w == 0) {
    float t = (ln < 8) ? buf[ln] : -3.0e38f;
#pragma unroll
    for (int off = 4; off; off >>= 1) t = fmaxf(t, __shfl_xor(t, off, 32));
    if (ln == 0) buf[0] = t;
  }
  __syncthreads();
  return buf[0];
}

__global__ __launch_bounds__(256) void k_layernorm(const float* __restrict__ x,
                                                   const float* __restrict__ g,
                                                   const float* __restrict__ bta,
                                                   __bf16* __restrict__ out) {
  __shared__ float buf[8];
  const int row = blockIdx.x;
  const float* xr = x + (size_t)row * kC;
  float s = 0.f, s2 = 0.f;
  for (int c = threadIdx.x; c < kC; c += 256) {
    const float v = xr[c];
    s += v;
    s2 += v * v;
  }
  const float S  = blk_sum(s, buf);
  const float S2 = blk_sum(s2, buf);
  const float mean = S / kC;
  const float rstd = rsqrtf(S2 / kC - mean * mean + 1e-5f);
  for (int c = threadIdx.x; c < kC; c += 256)
    out[(size_t)row * kC + c] = (__bf16)((xr[c] - mean) * rstd * g[c] + bta[c]);
}

__global__ void k_embed(const int* __restrict__ idx, const float* __restrict__ wte,
                        const float* __restrict__ wpe, float* __restrict__ x) {
  const int total = kM * kC;
  for (int i = blockIdx.x * blockDim.x + threadIdx.x; i < total;
       i += gridDim.x * blockDim.x) {
    const int r = i / kC, c = i - r * kC;
    const int t = r & (kT - 1);
    x[i] = wte[(size_t)idx[r] * kC + c] + wpe[(size_t)t * kC + c];
  }
}

// fp32 [l][K][N] -> bf16 transposed [l][N][K], LDS-tiled (coalesced both sides)
__global__ __launch_bounds__(256) void k_cast_t(const float* __restrict__ in,
                                                __bf16* __restrict__ out,
                                                int Kz, int Nz) {
  __shared__ float tile[32][33];
  const size_t lofs = (size_t)blockIdx.z * Kz * Nz;
  in += lofs;
  out += lofs;
  const int k0 = blockIdx.y * 32, n0 = blockIdx.x * 32;
  const int tx = threadIdx.x & 31, ty = threadIdx.x >> 5;  // 32 x 8
#pragma unroll
  for (int r = 0; r < 32; r += 8) {
    const int k = k0 + ty + r, n = n0 + tx;
    tile[ty + r][tx] = (k < Kz && n < Nz) ? in[(size_t)k * Nz + n] : 0.f;
  }
  __syncthreads();
#pragma unroll
  for (int r = 0; r < 32; r += 8) {
    const int n = n0 + ty + r, k = k0 + tx;
    if (n < Nz && k < Kz) out[(size_t)n * Kz + k] = (__bf16)tile[tx][ty + r];
  }
}

__global__ void k_zero(float* p) {
  if (threadIdx.x == 0 && blockIdx.x == 0) *p = 0.f;
}

__global__ __launch_bounds__(256) void k_loss(const float* __restrict__ logits,
                                              const int* __restrict__ tgt,
                                              float* __restrict__ loss) {
  __shared__ float buf[8];
  const int row = blockIdx.x;
  const float* lr = logits + (size_t)row * kV;
  float mx = -3.0e38f;
  for (int c = threadIdx.x; c < kV; c += 256) mx = fmaxf(mx, lr[c]);
  mx = blk_max(mx, buf);
  float se = 0.f;
  for (int c = threadIdx.x; c < kV; c += 256) se += expf(lr[c] - mx);
  se = blk_sum(se, buf);
  if (threadIdx.x == 0) {
    const float nll = logf(se) + mx - lr[tgt[row]];
    atomicAdd(loss, nll * (1.f / kM));
  }
}

// ---------------------------------------------------------------------------
// Host driver
// ---------------------------------------------------------------------------
extern "C" void kernel_launch(void* const* d_in, const int* in_sizes, int n_in,
                              void* d_out, int out_size, void* d_ws, size_t ws_size,
                              hipStream_t stream) {
  (void)in_sizes; (void)n_in; (void)out_size; (void)ws_size;

  const int*   idx         = (const int*)d_in[0];
  const int*   targets     = (const int*)d_in[1];
  const float* wte         = (const float*)d_in[2];
  const float* wpe         = (const float*)d_in[3];
  const float* ln1_g       = (const float*)d_in[4];
  const float* ln1_b       = (const float*)d_in[5];
  const float* attn_w      = (const float*)d_in[6];
  const float* attn_b      = (const float*)d_in[7];
  const float* attn_proj_w = (const float*)d_in[8];
  const float* attn_proj_b = (const float*)d_in[9];
  const float* ln2_g       = (const float*)d_in[10];
  const float* ln2_b       = (const float*)d_in[11];
  const float* fc_w        = (const float*)d_in[12];
  const float* fc_b        = (const float*)d_in[13];
  const float* mlp_proj_w  = (const float*)d_in[14];
  const float* mlp_proj_b  = (const float*)d_in[15];
  const float* lnf_g       = (const float*)d_in[16];
  const float* lnf_b       = (const float*)d_in[17];
  const float* lm_head_w   = (const float*)d_in[18];

  char* ws = (char*)d_ws;
  size_t off = 0;
  auto take = [&](size_t bytes) -> char* {
    char* p = ws + off;
    off = (off + bytes + 255) & ~(size_t)255;
    return p;
  };

  // all weights stored bf16 *transposed* [N,K] for async-friendly staging
  __bf16* w_qkv = (__bf16*)take((size_t)kL * kC * kC3 * 2);
  __bf16* w_ao  = (__bf16*)take((size_t)kL * kC * kC * 2);
  __bf16* w_fc  = (__bf16*)take((size_t)kL * kC * kF * 2);
  __bf16* w_mp  = (__bf16*)take((size_t)kL * kF * kC * 2);
  __bf16* w_lm  = (__bf16*)take((size_t)kC * kV * 2);
  float*  x     = (float*)take((size_t)kM * kC * 4);
  __bf16* hbuf  = (__bf16*)take((size_t)kM * kC * 2);
  float*  qkv   = (float*)take((size_t)kM * kC3 * 4);
  __bf16* ybuf  = (__bf16*)take((size_t)kM * kC * 2);
  __bf16* abuf  = (__bf16*)take((size_t)kM * kF * 2);

  const dim3 blk(256);

  // fp32 -> bf16 transposing casts (per layer: [K,N] -> [N,K])
  k_cast_t<<<dim3((kC3 + 31) / 32, (kC + 31) / 32, kL), blk, 0, stream>>>(
      attn_w, w_qkv, kC, kC3);
  k_cast_t<<<dim3((kC + 31) / 32, (kC + 31) / 32, kL), blk, 0, stream>>>(
      attn_proj_w, w_ao, kC, kC);
  k_cast_t<<<dim3((kF + 31) / 32, (kC + 31) / 32, kL), blk, 0, stream>>>(
      fc_w, w_fc, kC, kF);
  k_cast_t<<<dim3((kC + 31) / 32, (kF + 31) / 32, kL), blk, 0, stream>>>(
      mlp_proj_w, w_mp, kF, kC);
  k_cast_t<<<dim3((kV + 31) / 32, (kC + 31) / 32, 1), blk, 0, stream>>>(
      lm_head_w, w_lm, kC, kV);

  k_embed<<<2048, blk, 0, stream>>>(idx, wte, wpe, x);

  const dim3 gQKV((kC3 + 127) / 128, kM / 128);  // 18 x 16
  const dim3 gC((kC + 127) / 128, kM / 128);     // 6  x 16
  const dim3 gF((kF + 127) / 128, kM / 128);     // 24 x 16
  const dim3 gLM((kV + 127) / 128, kM / 128);    // 393 x 16
  const dim3 gAtt(kB * kH, kT / 128);            // 24 x 8

  for (int l = 0; l < kL; ++l) {
    k_layernorm<<<kM, blk, 0, stream>>>(x, ln1_g + (size_t)l * kC,
                                        ln1_b + (size_t)l * kC, hbuf);
    k_gemm<0, false><<<gQKV, blk, 0, stream>>>(
        hbuf, w_qkv + (size_t)l * kC * kC3, attn_b + (size_t)l * kC3, nullptr,
        qkv, nullptr, kM, kC3, kC);
    k_attn<<<gAtt, blk, 0, stream>>>(qkv, ybuf);
    k_gemm<0, true><<<gC, blk, 0, stream>>>(
        ybuf, w_ao + (size_t)l * kC * kC, attn_proj_b + (size_t)l * kC, x,
        x, nullptr, kM, kC, kC);
    k_layernorm<<<kM, blk, 0, stream>>>(x, ln2_g + (size_t)l * kC,
                                        ln2_b + (size_t)l * kC, hbuf);
    k_gemm<1, false><<<gF, blk, 0, stream>>>(
        hbuf, w_fc + (size_t)l * kC * kF, fc_b + (size_t)l * kF, nullptr,
        nullptr, abuf, kM, kF, kC);
    k_gemm<0, true><<<gC, blk, 0, stream>>>(
        abuf, w_mp + (size_t)l * kF * kC, mlp_proj_b + (size_t)l * kC, x,
        x, nullptr, kM, kC, kF);
  }

  k_layernorm<<<kM, blk, 0, stream>>>(x, lnf_g, lnf_b, hbuf);

  float* logits = (float*)d_out;
  k_gemm<0, false><<<gLM, blk, 0, stream>>>(hbuf, w_lm, nullptr, nullptr,
                                            logits, nullptr, kM, kV, kC);

  float* loss = logits + (size_t)kM * kV;
  k_zero<<<1, 32, 0, stream>>>(loss);
  k_loss<<<kM, blk, 0, stream>>>(logits, targets, loss);
}
